// ImageParallelModel_463856468076
// MI455X (gfx1250) — matrix operand order repaired and checked by reading, not compiled, not run on hardware
//
#include <hip/hip_runtime.h>
#include <math.h>

#define HH 512
#define WW 512
#define B_IN 224
#define B_OUT 420
#define N_LIB 480

#define N1_T 27   // phase-A col tiles (420 -> 432)
#define K1_T 7    // phase-A K chunks (224 / 32)
#define N2_T 30   // phase-B col tiles (480 / 16)
#define K2_T 14   // phase-B K chunks (448 / 32, 420 padded)
#define MROWS 4
#define PIX_PER_BLK (MROWS * 16)   // 64 pixels per block

typedef __attribute__((ext_vector_type(16))) __bf16 v16bf;
typedef __attribute__((ext_vector_type(8)))  float  v8f;

union Frag {
    unsigned int u[8];
    uint4        q[2];
    v16bf        v;
};

__device__ inline unsigned short f2bf(float f) {
    unsigned int u = __float_as_uint(f);
    if ((u & 0x7f800000u) == 0x7f800000u) return (unsigned short)(u >> 16); // inf/nan
    unsigned int r = u + 0x7fffu + ((u >> 16) & 1u);                         // RNE
    return (unsigned short)(r >> 16);
}
__device__ inline unsigned int pack2(float lo, float hi) {
    return (unsigned int)f2bf(lo) | ((unsigned int)f2bf(hi) << 16);
}

// ---------------------------------------------------------------------------
// Pre-pass: bf16-convert weights into WMMA B-fragment order.
// B fragment layout (bf16 32x16): lane = g*16 + n(0..15); dword i holds
// K = kc*32 + (i>>2)*16 + g*8 + (i&3)*2 (low half) and K+1 (high half).
// ---------------------------------------------------------------------------
__global__ void pack_R(const float* __restrict__ R, unsigned int* __restrict__ Rb) {
    int idx = blockIdx.x * blockDim.x + threadIdx.x;
    const int total = K1_T * N1_T * 256;
    if (idx >= total) return;
    int f = idx >> 8, rem = idx & 255;
    int lane = rem >> 3, i = rem & 7;
    int kc = f / N1_T, ct = f % N1_T;
    int g = lane >> 4;
    int k = kc * 32 + ((i >> 2) << 4) + (g << 3) + ((i & 3) << 1);
    int n = ct * 16 + (lane & 15);
    float lo = (n < B_OUT) ? R[n * B_IN + k]     : 0.f;
    float hi = (n < B_OUT) ? R[n * B_IN + k + 1] : 0.f;
    Rb[idx] = pack2(lo, hi);
}

__global__ void pack_A(const float* __restrict__ Am, unsigned int* __restrict__ Ab) {
    int idx = blockIdx.x * blockDim.x + threadIdx.x;
    const int total = K2_T * N2_T * 256;
    if (idx >= total) return;
    int f = idx >> 8, rem = idx & 255;
    int lane = rem >> 3, i = rem & 7;
    int kc = f / N2_T, ct = f % N2_T;
    int g = lane >> 4;
    int k = kc * 32 + ((i >> 2) << 4) + (g << 3) + ((i & 3) << 1);
    int n = ct * 16 + (lane & 15);
    float lo = (k     < B_OUT) ? Am[n * B_OUT + k]     : 0.f;
    float hi = (k + 1 < B_OUT) ? Am[n * B_OUT + k + 1] : 0.f;
    Ab[idx] = pack2(lo, hi);
}

// ---------------------------------------------------------------------------
// Fused: resample GEMM -> LDS -> library GEMM -> argmax -> angle.
// A fragment layout (bf16 16x32): lane = g*16 + m(0..15); dword i holds
// K = kc*32 + (i>>2)*16 + g*8 + (i&3)*2 (low half) and K+1 (high half).
// ---------------------------------------------------------------------------
__global__ __launch_bounds__(256) void fused_kernel(
        const float* __restrict__ data,
        const unsigned int* __restrict__ Rb,
        const unsigned int* __restrict__ Ab,
        const int* __restrict__ cls_in, const float* __restrict__ scr_in,
        int* __restrict__ cls_out, float* __restrict__ scr_out) {

    __shared__ __align__(16) unsigned int A_sm[MROWS * K1_T * 256];   // 28672 B
    __shared__ __align__(16) unsigned int Rs_sm[MROWS * K2_T * 256];  // 57344 B
    __shared__ float norm2[PIX_PER_BLK];
    __shared__ float bvw[8 * PIX_PER_BLK];
    __shared__ int   biw[8 * PIX_PER_BLK];

    const int tid   = threadIdx.x;
    const int wave  = tid >> 5;
    const int lane  = tid & 31;
    const int g     = lane >> 4;
    const int halfl = lane & 15;
    const int pixBase = blockIdx.x * PIX_PER_BLK;

    const v8f vzero = {0.f,0.f,0.f,0.f,0.f,0.f,0.f,0.f};

    // ---- stage image tile into LDS (A-fragment order), zero pads, init norms
    if (tid < PIX_PER_BLK) norm2[tid] = 0.f;
    for (int idx = tid; idx < MROWS * K2_T * 256; idx += 256) Rs_sm[idx] = 0u;
    for (int idx = tid; idx < MROWS * K1_T * 256; idx += 256) {
        int f = idx >> 8, rem = idx & 255;
        int l = rem >> 3, i = rem & 7;
        int t = f / K1_T, kc = f % K1_T;
        int gg = l >> 4;
        int k = kc * 32 + ((i >> 2) << 4) + (gg << 3) + ((i & 3) << 1);
        int row = t * 16 + (l & 15);
        const float* src = data + (size_t)(pixBase + row) * B_IN + k;
        A_sm[idx] = pack2(src[0], src[1]);
    }
    __syncthreads();

    // ---- phase A: resampled tile = data_tile x R^T  (bf16 WMMA, f32 acc)
    unsigned short* RsU = (unsigned short*)Rs_sm;
    for (int ct = wave; ct < N1_T; ct += 8) {
        v8f acc[MROWS];
        #pragma unroll
        for (int t = 0; t < MROWS; ++t) acc[t] = vzero;
        for (int kc = 0; kc < K1_T; ++kc) {
            Frag b;
            const uint4* bq = (const uint4*)(Rb + ((kc * N1_T + ct) * 256 + lane * 8));
            b.q[0] = bq[0]; b.q[1] = bq[1];
            #pragma unroll
            for (int t = 0; t < MROWS; ++t) {
                Frag a;
                const uint4* aq = (const uint4*)(A_sm + ((t * K1_T + kc) * 256 + lane * 8));
                a.q[0] = aq[0]; a.q[1] = aq[1];
                acc[t] = __builtin_amdgcn_wmma_f32_16x16x32_bf16(false, a.v, false, b.v,
                                                                 (short)0, acc[t], false, false);
            }
        }
        // scatter into phase-B A-fragment order + accumulate squared norms
        int N   = ct * 16 + halfl;
        int kc2 = N >> 5;
        int g2  = (N >> 3) & 1;
        int i2  = ((N >> 4) & 1) * 4 + ((N >> 1) & 3);
        int par = N & 1;
        #pragma unroll
        for (int t = 0; t < MROWS; ++t) {
            #pragma unroll
            for (int r = 0; r < 8; ++r) {
                int Mlo = r + (g << 3);
                float v = acc[t][r];
                RsU[(((t * K2_T + kc2) * 256 + (g2 * 16 + Mlo) * 8 + i2) << 1) + par] = f2bf(v);
                float sq = v * v;
                #pragma unroll
                for (int m = 1; m < 16; m <<= 1) sq += __shfl_xor(sq, m, 16);
                if (halfl == 0) atomicAdd(&norm2[t * 16 + Mlo], sq);
            }
        }
    }
    __syncthreads();

    // ---- phase B: dots vs library + running per-lane argmax
    float bv[MROWS][8]; int bi[MROWS][8];
    #pragma unroll
    for (int t = 0; t < MROWS; ++t)
        #pragma unroll
        for (int r = 0; r < 8; ++r) { bv[t][r] = -3.0e38f; bi[t][r] = 0; }

    for (int ct = wave; ct < N2_T; ct += 8) {
        v8f acc[MROWS];
        #pragma unroll
        for (int t = 0; t < MROWS; ++t) acc[t] = vzero;
        for (int kc = 0; kc < K2_T; ++kc) {
            Frag b;
            const uint4* bq = (const uint4*)(Ab + ((kc * N2_T + ct) * 256 + lane * 8));
            b.q[0] = bq[0]; b.q[1] = bq[1];
            #pragma unroll
            for (int t = 0; t < MROWS; ++t) {
                Frag a;
                const uint4* aq = (const uint4*)(Rs_sm + ((t * K2_T + kc) * 256 + lane * 8));
                a.q[0] = aq[0]; a.q[1] = aq[1];
                acc[t] = __builtin_amdgcn_wmma_f32_16x16x32_bf16(false, a.v, false, b.v,
                                                                 (short)0, acc[t], false, false);
            }
        }
        int n = ct * 16 + halfl;
        #pragma unroll
        for (int t = 0; t < MROWS; ++t)
            #pragma unroll
            for (int r = 0; r < 8; ++r)
                if (acc[t][r] > bv[t][r]) { bv[t][r] = acc[t][r]; bi[t][r] = n; }
    }

    // cross-lane (N) reduction within 16-lane halves, then stash per wave
    #pragma unroll
    for (int t = 0; t < MROWS; ++t) {
        #pragma unroll
        for (int r = 0; r < 8; ++r) {
            float v = bv[t][r]; int n = bi[t][r];
            #pragma unroll
            for (int m = 1; m < 16; m <<= 1) {
                float ov = __shfl_xor(v, m, 16); int on = __shfl_xor(n, m, 16);
                if (ov > v || (ov == v && on < n)) { v = ov; n = on; }
            }
            if (halfl == 0) {
                int M = t * 16 + r + (g << 3);
                bvw[wave * PIX_PER_BLK + M] = v;
                biw[wave * PIX_PER_BLK + M] = n;
            }
        }
    }
    __syncthreads();

    // ---- final per-pixel: cross-wave reduce, angle on winner only, mask
    if (tid < PIX_PER_BLK) {
        int m = tid;
        float bvbest = -3.0e38f; int bibest = 0;
        #pragma unroll
        for (int w = 0; w < 8; ++w) {
            float v = bvw[w * PIX_PER_BLK + m]; int n = biw[w * PIX_PER_BLK + m];
            if (v > bvbest || (v == bvbest && n < bibest)) { bvbest = v; bibest = n; }
        }
        float nrm = sqrtf(norm2[m]);
        float cv  = bvbest / fmaxf(nrm, 1e-12f);
        cv = fminf(1.f, fmaxf(-1.f, cv));
        float ang = 1.f - acosf(cv) * 0.3183098861837907f;   // 1/pi
        size_t p = (size_t)pixBase + m;
        float b0 = data[p * B_IN];
        bool valid = !((fabsf(b0 + 0.005f) <= (1e-8f + 1e-5f * 0.005f)) || (b0 == -50.0f));
        cls_out[p] = valid ? bibest : cls_in[p];
        scr_out[p] = valid ? ang    : scr_in[p];
    }
}

// ---------------------------------------------------------------------------
extern "C" void kernel_launch(void* const* d_in, const int* in_sizes, int n_in,
                              void* d_out, int out_size, void* d_ws, size_t ws_size,
                              hipStream_t stream) {
    const float* R    = (const float*)d_in[0];   // [420,224]
    const float* data = (const float*)d_in[1];   // [512,512,224]
    const float* Am   = (const float*)d_in[2];   // [480,420]
    const int*   cls  = (const int*)d_in[3];     // [512,512]
    const float* scr  = (const float*)d_in[4];   // [512,512]

    unsigned int* Rb = (unsigned int*)d_ws;                       // 193536 B
    unsigned int* Ab = Rb + (size_t)K1_T * N1_T * 256;            // 430080 B

    int*   cls_out = (int*)d_out;
    float* scr_out = (float*)d_out + (size_t)HH * WW;

    const int nR = K1_T * N1_T * 256;
    pack_R<<<(nR + 255) / 256, 256, 0, stream>>>(R, Rb);
    const int nA = K2_T * N2_T * 256;
    pack_A<<<(nA + 255) / 256, 256, 0, stream>>>(Am, Ab);

    fused_kernel<<<(HH * WW) / PIX_PER_BLK, 256, 0, stream>>>(
        data, Rb, Ab, cls, scr, cls_out, scr_out);
}
